// DelayedRNN_67705864454286
// MI455X (gfx1250) — matrix operand order repaired and checked by reading, not compiled, not run on hardware
//
#include <hip/hip_runtime.h>
#include <hip/hip_bf16.h>
#include <stdint.h>

// ---------------------------------------------------------------------------
// DelayedRNN for MI455X (gfx1250, wave32, WMMA).
// All matmuls run on v_wmma_f32_16x16x32_bf16 (fp32 accum). Weights are
// converted once to bf16 (L2-resident, ~18MB << 192MB L2). Recurrent scans
// are one fused kernel launch per step (GEMM + gates), column-partitioned
// across WGPs with double-buffered h for latency. Big GEMMs use LDS A-tiles;
// bf16 A-tiles use global_load_async_to_lds_b128.
// ---------------------------------------------------------------------------

typedef __bf16 bf16;
typedef __attribute__((ext_vector_type(16))) bf16  v16bf;
typedef __attribute__((ext_vector_type(8)))  float v8f;

typedef int v4i __attribute__((vector_size(16)));
typedef __attribute__((address_space(1))) v4i gv4i;  // global int4
typedef __attribute__((address_space(3))) v4i lv4i;  // LDS int4

#if defined(__HIP_DEVICE_COMPILE__) && __has_builtin(__builtin_amdgcn_global_load_async_to_lds_b128) && __has_builtin(__builtin_amdgcn_s_wait_asynccnt)
#define ASYNC_LDS 1
#else
#define ASYNC_LDS 0
#endif

static constexpr int kH = 512;
static constexpr int kB = 4;
static constexpr int kT = 256;
static constexpr int kF = 32;
static constexpr int kG = 1536;  // 3H

// ---------------------------------------------------------------- helpers --
__device__ inline v8f wmma_bf16(v16bf a, v16bf b, v8f c) {
  return __builtin_amdgcn_wmma_f32_16x16x32_bf16(false, a, false, b, (short)0, c,
                                                 false, false);
}

__device__ inline v16bf vzero16() {
  v16bf r;
#pragma unroll
  for (int i = 0; i < 16; ++i) r[i] = (bf16)0.0f;
  return r;
}

// A-fragment (16x32, 16-bit): lane group sel holds K {sel*16..+15}
__device__ inline v16bf load_afrag_bf16(const bf16* p /* row base + sel*16 */) {
  v16bf r;
#pragma unroll
  for (int i = 0; i < 8; ++i) { r[i] = p[i]; r[8 + i] = p[8 + i]; }
  return r;
}
__device__ inline v16bf load_afrag_f32(const float* p /* row base + sel*16 */) {
  v16bf r;
#pragma unroll
  for (int i = 0; i < 8; ++i) {
    r[i]     = (bf16)p[i];
    r[8 + i] = (bf16)p[8 + i];
  }
  return r;
}

__device__ inline float dsig(float x) { return 1.0f / (1.0f + __expf(-x)); }

// row -> element offset: r = (s, n), n = (nb, nq):  off = base + s*ss + nb*sb + nq*sq
__device__ inline long long rowoff(long long r, long long base, int Nb, int Fq,
                                   long long ss, long long sb, long long sq) {
  long long s = r / Nb;
  int n = (int)(r - s * Nb);
  return base + s * ss + (long long)(n / Fq) * sb + (long long)(n % Fq) * sq;
}

// ---------------------------------------------------------------- kernels --
__global__ void cvt_f32_bf16(const float* __restrict__ s, bf16* __restrict__ d,
                             long long n) {
  long long i = (long long)blockIdx.x * blockDim.x + threadIdx.x;
  long long st = (long long)gridDim.x * blockDim.x;
  for (; i < n; i += st) d[i] = (bf16)s[i];
}

__global__ void zero_f32(float* __restrict__ p, long long n) {
  long long i = (long long)blockIdx.x * blockDim.x + threadIdx.x;
  long long st = (long long)gridDim.x * blockDim.x;
  for (; i < n; i += st) p[i] = 0.0f;
}

// h_f_sum = input_h_f + output_h_t + output_h_c (broadcast over F), as bf16.
__global__ void hfsum_k(const float* __restrict__ inf_, const float* __restrict__ ot,
                        const float* __restrict__ oc, bf16* __restrict__ dst,
                        long long n) {
  long long i = (long long)blockIdx.x * blockDim.x + threadIdx.x;
  long long st = (long long)gridDim.x * blockDim.x;
  for (; i < n; i += st) {
    int hh = (int)(i & (kH - 1));
    long long bt = i >> 14;  // / (F*H) = 16384
    dst[i] = (bf16)(inf_[i] + ot[i] + oc[bt * kH + hh]);
  }
}

// Generic GEMM: out[r][j] = sum_k A(r,k) * W[j][k]  (+ bias[j]) (+ resid[r*N+j])
// A gathered via (base,Nb,Fq,ss,sb,sq) row map; W row-major [N][K] bf16.
// Tile 128x128, 256 threads = 8 waves (4x2), wave tile 32x64, K-step 32.
template <int ABF>
__global__ __launch_bounds__(256) void gemm_k(
    const void* __restrict__ Av, long long abase0, int Nb, int Fq,
    long long ss, long long sb, long long sq, const bf16* __restrict__ W,
    const float* __restrict__ bias, const float* __restrict__ resid,
    float* __restrict__ out, int M, int N, int K) {
  __shared__ alignas(64) bf16 As[128][32];

  const int tid = threadIdx.x;
  const int lane = tid & 31, wave = tid >> 5;
  const int wm = wave >> 1, wn = wave & 1;
  const int sel = lane >> 4, ln = lane & 15;

  // staging assignment: 256 threads cover 128 rows x 2 k-halves (16 elems each)
  const int tr = tid & 127, khalf = tid >> 7;
  const long long rs = (long long)blockIdx.x * 128 + tr;
  long long abase = -1;
  if (rs < M) abase = rowoff(rs, abase0, Nb, Fq, ss, sb, sq);

  v8f acc[2][4];
#pragma unroll
  for (int i = 0; i < 2; ++i)
#pragma unroll
    for (int j = 0; j < 4; ++j) acc[i][j] = (v8f)(0.0f);

  const int KC = K >> 5;
  for (int kc = 0; kc < KC; ++kc) {
    const int k0 = kc << 5;
    bf16* dst = &As[tr][khalf * 16];
    if (ABF) {
      const bf16* src = (const bf16*)Av + abase + k0 + khalf * 16;
#if ASYNC_LDS
      if (abase >= 0) {
        __builtin_amdgcn_global_load_async_to_lds_b128((gv4i*)src, (lv4i*)dst,
                                                       0, 0);
        __builtin_amdgcn_global_load_async_to_lds_b128((gv4i*)src, (lv4i*)dst,
                                                       16, 0);
      } else {
        *(v16bf*)dst = vzero16();
      }
#else
      if (abase >= 0) *(v16bf*)dst = *(const v16bf*)src;
      else            *(v16bf*)dst = vzero16();
#endif
    } else {
      if (abase >= 0) {
        const float* src = (const float*)Av + abase + k0 + khalf * 16;
        v16bf v;
#pragma unroll
        for (int i = 0; i < 16; ++i) v[i] = (bf16)src[i];
        *(v16bf*)dst = v;
      } else {
        *(v16bf*)dst = vzero16();
      }
    }
#if ASYNC_LDS
    if (ABF) __builtin_amdgcn_s_wait_asynccnt(0);
#endif
    __syncthreads();

    v16bf a0 = load_afrag_bf16(&As[wm * 32 + ln][sel * 16]);
    v16bf a1 = load_afrag_bf16(&As[wm * 32 + 16 + ln][sel * 16]);

    // prefetch next K-slice of this wave's weight columns into caches
    if (kc + 1 < KC)
      __builtin_prefetch(W + (long long)(blockIdx.y * 128 + wn * 64 + ln) * K +
                             k0 + 32, 0, 1);

#pragma unroll
    for (int ni = 0; ni < 4; ++ni) {
      const int j = blockIdx.y * 128 + wn * 64 + ni * 16 + ln;
      const bf16* bp = W + (long long)j * K + k0 + sel * 16;
      v16bf b = *(const v16bf*)bp;
      acc[0][ni] = wmma_bf16(a0, b, acc[0][ni]);
      acc[1][ni] = wmma_bf16(a1, b, acc[1][ni]);
    }
    __syncthreads();
  }

  // epilogue: C layout -> lane l: n = l%16, m = v + (l/16)*8
#pragma unroll
  for (int mi = 0; mi < 2; ++mi)
#pragma unroll
    for (int ni = 0; ni < 4; ++ni) {
      const int j = blockIdx.y * 128 + wn * 64 + ni * 16 + ln;
      const float bv = bias ? bias[j] : 0.0f;
#pragma unroll
      for (int vv = 0; vv < 8; ++vv) {
        const long long r =
            (long long)blockIdx.x * 128 + wm * 32 + mi * 16 + sel * 8 + vv;
        if (r < M) {
          const long long o = r * (long long)N + j;
          float val = acc[mi][ni][vv] + bv;
          if (resid) val += resid[o];
          out[o] = val;
        }
      }
    }
}

// One GRU scan step: gh = h @ w_hh^T + b_hh, gates, h_new.
// Grid: (ceil(Nb/16), 4 col-blocks of 128 hidden units). 256 threads = 8
// waves; wave w owns 16 hidden cols (jj = by*128 + w*16 + ln) in each of the
// r/z/n segments -> 3 accumulators, local gate combine, low reg pressure.
// h is double-buffered (hr = read, hw = write) so there are no hazards.
__global__ __launch_bounds__(256) void gru_step(
    const float* __restrict__ xg, const bf16* __restrict__ whh,
    const float* __restrict__ bhh, const float* __restrict__ hr,
    float* __restrict__ hw, bf16* __restrict__ outp, int s, int Nb, int Fq,
    long long ob, long long oss, long long osb, long long osq) {
  const int tid = threadIdx.x;
  const int lane = tid & 31, wave = tid >> 5;
  const int sel = lane >> 4, ln = lane & 15;
  const int row_a = blockIdx.x * 16 + ln;       // A-fragment row for this lane
  const int jj0 = blockIdx.y * 128 + wave * 16; // this wave's hidden col base

  v8f acc[3];
#pragma unroll
  for (int i = 0; i < 3; ++i) acc[i] = (v8f)(0.0f);

  const int jcol = jj0 + ln;  // B column for this lane (per segment +512)
  for (int kc = 0; kc < 16; ++kc) {
    v16bf a;
    if (row_a < Nb)
      a = load_afrag_f32(hr + (long long)row_a * kH + kc * 32 + sel * 16);
    else
      a = vzero16();
    const bf16* bbase = whh + kc * 32 + sel * 16;
    // three independent loads in flight, then three WMMAs
    v16bf b0 = *(const v16bf*)(bbase + (long long)(jcol) * kH);
    v16bf b1 = *(const v16bf*)(bbase + (long long)(jcol + kH) * kH);
    v16bf b2 = *(const v16bf*)(bbase + (long long)(jcol + 2 * kH) * kH);
    acc[0] = wmma_bf16(a, b0, acc[0]);
    acc[1] = wmma_bf16(a, b1, acc[1]);
    acc[2] = wmma_bf16(a, b2, acc[2]);
  }

  // gates + state update (C layout: lane l -> n = l%16, m = v + (l/16)*8)
  const int jj = jj0 + ln;
  const float br = bhh[jj], bz = bhh[jj + kH], bn = bhh[jj + 2 * kH];
#pragma unroll
  for (int vv = 0; vv < 8; ++vv) {
    const int row = blockIdx.x * 16 + sel * 8 + vv;
    if (row < Nb) {
      const float* xr = xg + ((long long)s * Nb + row) * kG;
      const float r = dsig(xr[jj] + acc[0][vv] + br);
      const float z = dsig(xr[jj + kH] + acc[1][vv] + bz);
      const float nn = tanhf(xr[jj + 2 * kH] + r * (acc[2][vv] + bn));
      const float hn = (1.0f - z) * nn + z * hr[(long long)row * kH + jj];
      hw[(long long)row * kH + jj] = hn;
      const long long oo = ob + (long long)s * oss +
                           (long long)(row / Fq) * osb +
                           (long long)(row % Fq) * osq + jj;
      outp[oo] = (bf16)hn;
    }
  }
}

// ------------------------------------------------------------------- host --
extern "C" void kernel_launch(void* const* d_in, const int* in_sizes, int n_in,
                              void* d_out, int out_size, void* d_ws,
                              size_t ws_size, hipStream_t stream) {
  (void)in_sizes; (void)n_in; (void)out_size; (void)ws_size;

  const float* in_t = (const float*)d_in[0];
  const float* in_f = (const float*)d_in[1];
  const float* in_c = (const float*)d_in[2];
  const float *wih[5], *whh[5], *bih[5], *bhh[5];
  for (int g = 0; g < 5; ++g) {  // tx, ty, tz, c, f
    wih[g] = (const float*)d_in[3 + 4 * g];
    whh[g] = (const float*)d_in[4 + 4 * g];
    bih[g] = (const float*)d_in[5 + 4 * g];
    bhh[g] = (const float*)d_in[6 + 4 * g];
  }
  const float* W_t = (const float*)d_in[23]; const float* b_t = (const float*)d_in[24];
  const float* W_c = (const float*)d_in[25]; const float* b_c = (const float*)d_in[26];
  const float* W_f = (const float*)d_in[27]; const float* b_f = (const float*)d_in[28];

  const long long NE = (long long)kB * kT * kF * kH;  // 16,777,216
  float* out_t = (float*)d_out;
  float* out_f = out_t + NE;
  float* out_c = out_f + NE;

  char* w = (char*)d_ws;
  auto take = [&](size_t bytes) {
    char* p = w;
    w += (bytes + 255) & ~(size_t)255;
    return p;
  };
  bf16 *wihb[5], *whhb[5];
  for (int g = 0; g < 5; ++g) {
    wihb[g] = (bf16*)take((size_t)kG * kH * 2);
    whhb[g] = (bf16*)take((size_t)kG * kH * 2);
  }
  bf16* Wtb = (bf16*)take((size_t)kH * kG * 2);
  bf16* Wcb = (bf16*)take((size_t)kH * kH * 2);
  bf16* Wfb = (bf16*)take((size_t)kH * kH * 2);
  float* xg    = (float*)take((size_t)32768 * kG * 4);  // reused per GRU
  float* hbuf0 = (float*)take((size_t)1024 * kH * 4);   // h ping
  float* hbuf1 = (float*)take((size_t)1024 * kH * 4);   // h pong
  bf16* concat = (bf16*)take((size_t)32768 * kG * 2);   // [B,T,F,3H]
  bf16* hfs    = (bf16*)take((size_t)NE * 2);           // h_f_sum   [B,T,F,H]
  bf16* hft    = (bf16*)take((size_t)NE * 2);           // h_f_temp  [B,T,F,H]
  bf16* hct    = (bf16*)take((size_t)1024 * kH * 2);    // h_c_temp  [B,T,H]

  auto cvt = [&](const float* s, bf16* d, long long n) {
    int blocks = (int)((n + 256 * 16 - 1) / (256 * 16));
    if (blocks > 4096) blocks = 4096;
    cvt_f32_bf16<<<blocks, 256, 0, stream>>>(s, d, n);
  };
  for (int g = 0; g < 5; ++g) {
    cvt(wih[g], wihb[g], (long long)kG * kH);
    cvt(whh[g], whhb[g], (long long)kG * kH);
  }
  cvt(W_t, Wtb, (long long)kH * kG);
  cvt(W_c, Wcb, (long long)kH * kH);
  cvt(W_f, Wfb, (long long)kH * kH);

  auto gemmA32 = [&](const float* A, long long base, int Nb, int Fq,
                     long long ss, long long sb, long long sq, const bf16* Wm,
                     const float* bias, const float* resid, float* out, int M,
                     int N, int K) {
    gemm_k<0><<<dim3((M + 127) / 128, N / 128), 256, 0, stream>>>(
        (const void*)A, base, Nb, Fq, ss, sb, sq, Wm, bias, resid, out, M, N, K);
  };
  auto gemmA16 = [&](const bf16* A, long long base, int Nb, int Fq,
                     long long ss, long long sb, long long sq, const bf16* Wm,
                     const float* bias, const float* resid, float* out, int M,
                     int N, int K) {
    gemm_k<1><<<dim3((M + 127) / 128, N / 128), 256, 0, stream>>>(
        (const void*)A, base, Nb, Fq, ss, sb, sq, Wm, bias, resid, out, M, N, K);
  };
  auto scan = [&](const bf16* whhb_, const float* bhh_, bf16* outp, int Nb,
                  int S, int Fq, long long ob, long long oss, long long osb,
                  long long osq) {
    zero_f32<<<512, 256, 0, stream>>>(hbuf0, (long long)Nb * kH);
    const int wgs = (Nb + 15) / 16;
    float* ha = hbuf0;  // read
    float* hb = hbuf1;  // write
    for (int s = 0; s < S; ++s) {
      gru_step<<<dim3(wgs, kH / 128), 256, 0, stream>>>(
          xg, whhb_, bhh_, ha, hb, outp, s, Nb, Fq, ob, oss, osb, osq);
      float* t = ha; ha = hb; hb = t;
    }
  };

  const long long TFH = (long long)kT * kF * kH;   // 4,194,304
  const long long TFG = (long long)kT * kF * kG;   // 12,582,912
  const long long FH = (long long)kF * kH;         // 16,384
  const long long FG = (long long)kF * kG;         // 49,152

  // ---- GRU tx: scan over T, rows n = b*F+f, S=256, Nb=128 -----------------
  gemmA32(in_t, 0, 128, kF, FH, TFH, kH, wihb[0], bih[0], nullptr, xg,
          32768, kG, kH);
  scan(whhb[0], bhh[0], concat, 128, kT, kF, /*ob=*/0, /*oss=*/FG,
       /*osb=*/TFG, /*osq=*/kG);

  // ---- GRU ty: scan over F (fwd), rows n = b*T+t, S=32, Nb=1024 -----------
  gemmA32(in_t, 0, 1024, kT, kH, TFH, FH, wihb[1], bih[1], nullptr, xg,
          32768, kG, kH);
  scan(whhb[1], bhh[1], concat, 1024, kF, kT, /*ob=*/kH, /*oss=*/kG,
       /*osb=*/TFG, /*osq=*/FG);

  // ---- GRU tz: scan over F reversed ---------------------------------------
  gemmA32(in_t, (long long)(kF - 1) * kH, 1024, kT, -(long long)kH, TFH, FH,
          wihb[2], bih[2], nullptr, xg, 32768, kG, kH);
  scan(whhb[2], bhh[2], concat, 1024, kF, kT,
       /*ob=*/2 * kH + (long long)(kF - 1) * kG, /*oss=*/-(long long)kG,
       /*osb=*/TFG, /*osq=*/FG);

  // ---- GRU c: scan over T, rows n = b, S=256, Nb=4 ------------------------
  gemmA32(in_c, 0, 4, 1, kH, (long long)kT * kH, 0, wihb[3], bih[3], nullptr,
          xg, 1024, kG, kH);
  scan(whhb[3], bhh[3], hct, 4, kT, 1, /*ob=*/0, /*oss=*/kH,
       /*osb=*/(long long)kT * kH, /*osq=*/0);

  // output_h_c = input_h_c + h_c_temp @ W_c^T + b_c
  gemmA16(hct, 0, 1, 1, kH, 0, 0, Wcb, b_c, in_c, out_c, 1024, kH, kH);

  // output_h_t = input_h_t + concat @ W_t^T + b_t
  gemmA16(concat, 0, 1, 1, kG, 0, 0, Wtb, b_t, in_t, out_t, 32768, kH, kG);

  // h_f_sum = input_h_f + output_h_t + output_h_c (broadcast)
  hfsum_k<<<8192, 256, 0, stream>>>(in_f, out_t, out_c, hfs, NE);

  // ---- GRU f: scan over F, rows n = b*T+t, input = h_f_sum (bf16) ---------
  gemmA16(hfs, 0, 1024, kT, kH, TFH, FH, wihb[4], bih[4], nullptr, xg,
          32768, kG, kH);
  scan(whhb[4], bhh[4], hft, 1024, kF, kT, /*ob=*/0, /*oss=*/kH,
       /*osb=*/TFH, /*osq=*/FH);

  // output_h_f = input_h_f + h_f_temp @ W_f^T + b_f
  gemmA16(hft, 0, 1, 1, kH, 0, 0, Wfb, b_f, in_f, out_f, 32768, kH, kH);
}
